// BigramLM_19258633355338
// MI455X (gfx1250) — compile-verified, matrix operand order checked by reading, not run on hardware
//
#include <hip/hip_runtime.h>
#include <hip/hip_bf16.h>
#include <math.h>

typedef __bf16 bf16;
typedef __attribute__((ext_vector_type(16))) __bf16 v16bf;
typedef __attribute__((ext_vector_type(8)))  __bf16 v8bf;
typedef __attribute__((ext_vector_type(8)))  float  v8f;

#define WPB  4  // wave32 waves per block
#define NSUB 4  // 16x16 sub-tiles per wave -> each wave owns a 16x64 C strip

__device__ __forceinline__ v16bf load_frag(const bf16* p)
{
  // element j<8 -> K = +j ; j>=8 -> K = +16+(j-8)   (lane already offset by 8*lh)
  v8bf lo = *(const v8bf*)(p);
  v8bf hi = *(const v8bf*)(p + 16);
  return __builtin_shufflevector(lo, hi, 0, 1, 2, 3, 4, 5, 6, 7,
                                 8, 9, 10, 11, 12, 13, 14, 15);
}

// ---------------------------------------------------------------------------
// bf16 WMMA GEMM: C[z][M,N] = scale * A[z][M,K] x B[z][K,N] + bias
// B is passed TRANSPOSED (Bt, row-major [N,K], row stride ldb) so every lane's
// operand fragment is two contiguous 16-byte chunks -> global_load_b128 pairs.
// Each wave computes a 16x(16*NSUB) strip: A fragment loaded once per K-step,
// reused against NSUB B fragments (2.5 loads per WMMA instead of 4). All
// fragment loads for a K-step are issued before the first WMMA so the
// scheduler can use partial s_wait_loadcnt and overlap loads with compute.
// The K loop is fully branchless: out-of-range columns are clamped to a valid
// address, computed, and simply never stored; causal skipping is only done at
// whole-strip granularity (wave-uniform early return).
// OUT_MODE: 0 = f32 store, 1 = f32 residual add (+=), 2 = bf16 store
// ---------------------------------------------------------------------------
template <int OUT_MODE, bool RELU, bool CAUSAL>
__global__ __launch_bounds__(32 * WPB) void gemm_wmma(
    const bf16* __restrict__ A, long long sAzb, long long sAzh, int lda,
    const bf16* __restrict__ Bt, long long sBzb, long long sBzh, int ldb,
    void* __restrict__ Cp, long long sCzb, long long sCzh, int ldc,
    const float* __restrict__ bias,
    int M, int N, int K, int ZH, float scale)
{
  int z  = blockIdx.z;
  int zb = z / ZH;
  int zh = z - zb * ZH;
  const bf16* Ab = A  + zb * sAzb + zh * sAzh;
  const bf16* Bb = Bt + zb * sBzb + zh * sBzh;

  int wave = threadIdx.x >> 5;
  int lane = threadIdx.x & 31;
  int lh   = lane >> 4;   // half-wave: selects K sub-block
  int ln16 = lane & 15;

  int n0 = (blockIdx.x * WPB + wave) * (16 * NSUB);
  int m0 = blockIdx.y * 16;
  if (n0 >= N) return;                  // wave-uniform
  if (CAUSAL && n0 >= m0 + 16) return;  // strip fully above causal diagonal

  const bf16* arow = Ab + (size_t)(m0 + ln16) * lda + 8 * lh;
  const bf16* brow[NSUB];
#pragma unroll
  for (int i = 0; i < NSUB; ++i) {
    int n  = n0 + 16 * i + ln16;
    int nc = n < N ? n : N - 1;  // clamp: extra columns computed, never stored
    brow[i] = Bb + (size_t)nc * ldb + 8 * lh;
  }

  v8f acc[NSUB] = {};
  for (int k0 = 0; k0 < K; k0 += 32) {
    // issue every load of this K-step before the first WMMA
    v16bf av = load_frag(arow + k0);
    v16bf bv[NSUB];
#pragma unroll
    for (int i = 0; i < NSUB; ++i)
      bv[i] = load_frag(brow[i] + k0);
#pragma unroll
    for (int i = 0; i < NSUB; ++i)
      acc[i] = __builtin_amdgcn_wmma_f32_16x16x32_bf16(false, av, false, bv[i],
                                                       (short)0, acc[i], false, false);
  }

#pragma unroll
  for (int i = 0; i < NSUB; ++i) {
    int n = n0 + 16 * i + ln16;
    if (n >= N) continue;  // store-side guard only
    float bs = bias ? bias[n] : 0.0f;

    if (OUT_MODE == 2) {
      bf16* C = (bf16*)Cp + zb * sCzb + zh * sCzh;
#pragma unroll
      for (int r = 0; r < 8; ++r) {
        int   m = m0 + r + 8 * lh;
        float v = acc[i][r] * scale + bs;
        if (RELU) v = v > 0.0f ? v : 0.0f;
        C[(size_t)m * ldc + n] = (bf16)v;
      }
    } else if (OUT_MODE == 1) {
      float* C = (float*)Cp + zb * sCzb + zh * sCzh;
#pragma unroll
      for (int r = 0; r < 8; ++r) {
        int m = m0 + r + 8 * lh;
        C[(size_t)m * ldc + n] += acc[i][r] * scale + bs;
      }
    } else {
      float* C = (float*)Cp + zb * sCzb + zh * sCzh;
#pragma unroll
      for (int r = 0; r < 8; ++r) {
        int   m = m0 + r + 8 * lh;
        float v = acc[i][r] * scale + bs;
        if (RELU) v = v > 0.0f ? v : 0.0f;
        C[(size_t)m * ldc + n] = v;
      }
    }
  }
}

// ---------------------------------------------------------------------------
// x = tok_emb[idx] + pos_emb
// ---------------------------------------------------------------------------
__global__ __launch_bounds__(256) void embed_kernel(
    const int* __restrict__ idx, const float* __restrict__ tok,
    const float* __restrict__ pos, float* __restrict__ X, int Dm, int T)
{
  int row = blockIdx.x;
  int tk  = idx[row];
  int t   = row & (T - 1);  // row = b*T + t, T power of 2
  for (int d = threadIdx.x; d < Dm; d += 256)
    X[(size_t)row * Dm + d] = tok[(size_t)tk * Dm + d] + pos[(size_t)t * Dm + d];
}

// ---------------------------------------------------------------------------
// LayerNorm row -> bf16
// ---------------------------------------------------------------------------
__global__ __launch_bounds__(256) void ln_kernel(
    const float* __restrict__ X, const float* __restrict__ g,
    const float* __restrict__ b, bf16* __restrict__ out, int Dm)
{
  int row = blockIdx.x;
  const float* x = X + (size_t)row * Dm;
  __shared__ float s1[256], s2[256];
  float ls = 0.0f, lq = 0.0f;
  for (int d = threadIdx.x; d < Dm; d += 256) {
    float v = x[d];
    ls += v;
    lq += v * v;
  }
  s1[threadIdx.x] = ls;
  s2[threadIdx.x] = lq;
  __syncthreads();
  for (int o = 128; o > 0; o >>= 1) {
    if (threadIdx.x < o) {
      s1[threadIdx.x] += s1[threadIdx.x + o];
      s2[threadIdx.x] += s2[threadIdx.x + o];
    }
    __syncthreads();
  }
  float mu  = s1[0] / Dm;
  float var = s2[0] / Dm - mu * mu;
  float rs  = rsqrtf(var + 1e-5f);
  bf16* op  = out + (size_t)row * Dm;
  for (int d = threadIdx.x; d < Dm; d += 256)
    op[d] = (bf16)((x[d] - mu) * rs * g[d] + b[d]);
}

// ---------------------------------------------------------------------------
// In-place causal softmax over a score row (bf16). Writes zeros past t so the
// following P*V GEMM needs no masking. Each thread only touches its own
// strided element subset -> safe in place.
// ---------------------------------------------------------------------------
__global__ __launch_bounds__(256) void softmax_kernel(bf16* __restrict__ S, int T)
{
  int   t   = blockIdx.x;
  int   z   = blockIdx.y;
  bf16* row = S + (size_t)z * T * T + (size_t)t * T;
  int   len = t + 1;
  __shared__ float red[256];

  float lmax = -3.0e38f;
  for (int s = threadIdx.x; s < len; s += 256)
    lmax = fmaxf(lmax, (float)row[s]);
  red[threadIdx.x] = lmax;
  __syncthreads();
  for (int o = 128; o > 0; o >>= 1) {
    if (threadIdx.x < o) red[threadIdx.x] = fmaxf(red[threadIdx.x], red[threadIdx.x + o]);
    __syncthreads();
  }
  float mx = red[0];
  __syncthreads();

  float lsum = 0.0f;
  for (int s = threadIdx.x; s < len; s += 256)
    lsum += __expf((float)row[s] - mx);
  red[threadIdx.x] = lsum;
  __syncthreads();
  for (int o = 128; o > 0; o >>= 1) {
    if (threadIdx.x < o) red[threadIdx.x] += red[threadIdx.x + o];
    __syncthreads();
  }
  float inv = 1.0f / red[0];

  for (int s = threadIdx.x; s < T; s += 256) {
    float p = (s < len) ? __expf((float)row[s] - mx) * inv : 0.0f;
    row[s]  = (bf16)p;
  }
}

// ---------------------------------------------------------------------------
// Weight conversion + transpose: out[n*K + k] = (bf16)in[k*N + n]
// (coalesced writes; strided reads served by L2)
// ---------------------------------------------------------------------------
__global__ __launch_bounds__(256) void f2bf_t_kernel(
    const float* __restrict__ in, bf16* __restrict__ out, int K, int N)
{
  long long o = (long long)blockIdx.x * 256 + threadIdx.x;
  if (o >= (long long)K * N) return;
  int n = (int)(o / K);
  int k = (int)(o - (long long)n * K);
  out[o] = (bf16)in[(size_t)k * N + n];
}

// Wk (H, D, HS) -> transposed bf16 [H*HS, D]: WB[(h*HS+e)*D + d] = Wk[h][d][e]
__global__ __launch_bounds__(256) void wk_convert_kernel(
    const float* __restrict__ Wk, bf16* __restrict__ WB, int Hh, int Dm, int HS)
{
  int i   = blockIdx.x * 256 + threadIdx.x;
  int tot = Hh * Dm * HS;
  if (i >= tot) return;
  int h = i / (Dm * HS);
  int r = i - h * Dm * HS;
  int d = r / HS;
  int e = r - d * HS;
  WB[(size_t)(h * HS + e) * Dm + d] = (bf16)Wk[i];
}

// K [b,t,h,e] -> Kt [b,h,e,t] so the P*V GEMM's B is K-contiguous
__global__ __launch_bounds__(256) void kt_transpose_kernel(
    const bf16* __restrict__ Kb, bf16* __restrict__ Kt,
    int Bn, int T, int Hh, int HS)
{
  long long o   = (long long)blockIdx.x * 256 + threadIdx.x;
  long long tot = (long long)Bn * Hh * HS * T;
  if (o >= tot) return;
  int s  = (int)(o % T);
  long long r1 = o / T;
  int e  = (int)(r1 % HS);
  long long r2 = r1 / HS;
  int h  = (int)(r2 % Hh);
  int b  = (int)(r2 / Hh);
  Kt[o] = Kb[((size_t)(b * T + s)) * (Hh * HS) + h * HS + e];
}

__global__ void zero_kernel(float* p) { *p = 0.0f; }

// ---------------------------------------------------------------------------
// Cross-entropy: loss += -(logit[tgt] - max - log(sum exp)) / rows
// ---------------------------------------------------------------------------
__global__ __launch_bounds__(256) void loss_kernel(
    const float* __restrict__ logits, const int* __restrict__ targets,
    float* __restrict__ loss, int V, int rows)
{
  int row = blockIdx.x;
  const float* x = logits + (size_t)row * V;
  __shared__ float red[256];

  float lmax = -3.0e38f;
  for (int i = threadIdx.x; i < V; i += 256) lmax = fmaxf(lmax, x[i]);
  red[threadIdx.x] = lmax;
  __syncthreads();
  for (int o = 128; o > 0; o >>= 1) {
    if (threadIdx.x < o) red[threadIdx.x] = fmaxf(red[threadIdx.x], red[threadIdx.x + o]);
    __syncthreads();
  }
  float mx = red[0];
  __syncthreads();

  float ls = 0.0f;
  for (int i = threadIdx.x; i < V; i += 256) ls += __expf(x[i] - mx);
  red[threadIdx.x] = ls;
  __syncthreads();
  for (int o = 128; o > 0; o >>= 1) {
    if (threadIdx.x < o) red[threadIdx.x] += red[threadIdx.x + o];
    __syncthreads();
  }
  if (threadIdx.x == 0) {
    float lp = x[targets[row]] - mx - logf(red[0]);
    atomicAdd(loss, -lp / (float)rows);
  }
}

// ---------------------------------------------------------------------------
static inline dim3 ggrid(int M, int N, int Z)
{
  int strips = (N + 16 * NSUB - 1) / (16 * NSUB);
  return dim3((strips + WPB - 1) / WPB, M / 16, Z);
}

extern "C" void kernel_launch(void* const* d_in, const int* in_sizes, int n_in,
                              void* d_out, int out_size, void* d_ws, size_t ws_size,
                              hipStream_t stream)
{
  (void)in_sizes; (void)n_in; (void)out_size; (void)ws_size;

  const int*   idx   = (const int*)d_in[0];
  const int*   tgt   = (const int*)d_in[1];
  const float* tok   = (const float*)d_in[2];
  const float* pos   = (const float*)d_in[3];
  const float* Wk    = (const float*)d_in[4];
  const float* Wproj = (const float*)d_in[5];
  const float* bproj = (const float*)d_in[6];
  const float* ln1g  = (const float*)d_in[7];
  const float* ln1b  = (const float*)d_in[8];
  const float* ln2g  = (const float*)d_in[9];
  const float* ln2b  = (const float*)d_in[10];
  const float* W1    = (const float*)d_in[11];
  const float* b1    = (const float*)d_in[12];
  const float* W2    = (const float*)d_in[13];
  const float* b2    = (const float*)d_in[14];
  const float* lnfg  = (const float*)d_in[15];
  const float* lnfb  = (const float*)d_in[16];
  const float* Wlm   = (const float*)d_in[17];
  const float* blm   = (const float*)d_in[18];

  constexpr int Bn = 4, T = 1024, Dm = 768, Hh = 12, Ll = 6, HS = 64;
  constexpr int V = 50257, F = 3072;
  constexpr int BT = Bn * T;  // 4096

  // workspace carve-out (256B aligned)
  char* wsp = (char*)d_ws;
  auto carve = [&](size_t bytes) -> char* {
    char* p = wsp;
    wsp += (bytes + 255) & ~(size_t)255;
    return p;
  };
  float* X  = (float*)carve((size_t)BT * Dm * 4);        // residual stream
  bf16*  Hb = (bf16*)carve((size_t)BT * Dm * 2);         // LN output
  bf16*  Kb = (bf16*)carve((size_t)BT * Dm * 2);         // k (= q = v) [b,t,h,e]
  bf16*  Kt = (bf16*)carve((size_t)BT * Dm * 2);         // K^T [b,h,e,t]
  bf16*  Ob = (bf16*)carve((size_t)BT * Dm * 2);         // attention out
  bf16*  Sb = (bf16*)carve((size_t)Bn * Hh * T * T * 2); // scores / probs
  bf16*  A1 = (bf16*)carve((size_t)BT * F * 2);          // MLP hidden
  bf16*  WB = (bf16*)carve((size_t)Dm * V * 2);          // converted weights (B^T)

  float* logits = (float*)d_out;
  float* loss   = logits + (size_t)BT * V;

  const dim3  blk(32 * WPB);
  const float scl = 1.0f / sqrtf((float)Dm);  // reference scales by D**-0.5

  embed_kernel<<<BT, 256, 0, stream>>>(idx, tok, pos, X, Dm, T);
  zero_kernel<<<1, 1, 0, stream>>>(loss);

  for (int l = 0; l < Ll; ++l) {
    // --- attention ---
    ln_kernel<<<BT, 256, 0, stream>>>(X, ln1g + (size_t)l * Dm, ln1b + (size_t)l * Dm, Hb, Dm);

    {
      int tot = Hh * Dm * HS;
      wk_convert_kernel<<<(tot + 255) / 256, 256, 0, stream>>>(
          Wk + (size_t)l * tot, WB, Hh, Dm, HS);
    }
    // K = Hb @ Wk   [BT, Dm] (B^T layout, ldb = Dm)
    gemm_wmma<2, false, false><<<ggrid(BT, Dm, 1), blk, 0, stream>>>(
        Hb, 0, 0, Dm, WB, 0, 0, Dm, Kb, 0, 0, Dm,
        nullptr, BT, Dm, Dm, 1, 1.0f);

    // K^T copy for the P*V GEMM
    {
      long long tot = (long long)BT * Dm;
      kt_transpose_kernel<<<(unsigned)((tot + 255) / 256), 256, 0, stream>>>(
          Kb, Kt, Bn, T, Hh, HS);
    }

    // S[b,h] = scale * K K^T (strips above causal diagonal skipped)
    gemm_wmma<2, false, true><<<ggrid(T, T, Bn * Hh), blk, 0, stream>>>(
        Kb, (long long)T * Dm, HS, Dm,
        Kb, (long long)T * Dm, HS, Dm,
        Sb, (long long)Hh * T * T, (long long)T * T, T,
        nullptr, T, T, HS, Hh, scl);

    softmax_kernel<<<dim3(T, Bn * Hh), 256, 0, stream>>>(Sb, T);

    // O[b,t,h,e] = P @ K   (B^T = Kt, ldb = T)
    gemm_wmma<2, false, false><<<ggrid(T, HS, Bn * Hh), blk, 0, stream>>>(
        Sb, (long long)Hh * T * T, (long long)T * T, T,
        Kt, (long long)Hh * HS * T, (long long)HS * T, T,
        Ob, (long long)T * Dm, HS, Dm,
        nullptr, T, HS, T, Hh, 1.0f);

    // X += O @ Wproj + bproj
    {
      long long n = (long long)Dm * Dm;
      f2bf_t_kernel<<<(unsigned)((n + 255) / 256), 256, 0, stream>>>(
          Wproj + (size_t)l * Dm * Dm, WB, Dm, Dm);
    }
    gemm_wmma<1, false, false><<<ggrid(BT, Dm, 1), blk, 0, stream>>>(
        Ob, 0, 0, Dm, WB, 0, 0, Dm, X, 0, 0, Dm,
        bproj + (size_t)l * Dm, BT, Dm, Dm, 1, 1.0f);

    // --- MLP ---
    ln_kernel<<<BT, 256, 0, stream>>>(X, ln2g + (size_t)l * Dm, ln2b + (size_t)l * Dm, Hb, Dm);

    {
      long long n = (long long)Dm * F;
      f2bf_t_kernel<<<(unsigned)((n + 255) / 256), 256, 0, stream>>>(
          W1 + (size_t)l * Dm * F, WB, Dm, F);
    }
    gemm_wmma<2, true, false><<<ggrid(BT, F, 1), blk, 0, stream>>>(
        Hb, 0, 0, Dm, WB, 0, 0, Dm, A1, 0, 0, F,
        b1 + (size_t)l * F, BT, F, Dm, 1, 1.0f);

    {
      long long n = (long long)F * Dm;
      f2bf_t_kernel<<<(unsigned)((n + 255) / 256), 256, 0, stream>>>(
          W2 + (size_t)l * F * Dm, WB, F, Dm);
    }
    gemm_wmma<1, false, false><<<ggrid(BT, Dm, 1), blk, 0, stream>>>(
        A1, 0, 0, F, WB, 0, 0, F, X, 0, 0, Dm,
        b2 + (size_t)l * Dm, BT, Dm, F, 1, 1.0f);
  }

  // --- final LN + LM head ---
  ln_kernel<<<BT, 256, 0, stream>>>(X, lnfg, lnfb, Hb, Dm);
  {
    long long n = (long long)Dm * V;
    f2bf_t_kernel<<<(unsigned)((n + 255) / 256), 256, 0, stream>>>(Wlm, WB, Dm, V);
  }
  gemm_wmma<0, false, false><<<ggrid(BT, V, 1), blk, 0, stream>>>(
      Hb, 0, 0, Dm, WB, 0, 0, Dm, logits, 0, 0, V,
      blm, BT, V, Dm, 1, 1.0f);

  loss_kernel<<<BT, 256, 0, stream>>>(logits, tgt, loss, V, BT);
}
